// CFDSurrogateModel_62440234549306
// MI455X (gfx1250) — matrix operand order repaired and checked by reading, not compiled, not run on hardware
//
#include <hip/hip_runtime.h>
#include <hip/hip_bf16.h>

// ---------------------------------------------------------------------------
// MeshGraphNet-style CFD surrogate on gfx1250 (MI455X).
// Mixed precision: fp32 residual/LN state + bf16 shadow activations feeding
// v_wmma_f32_16x16x32_bf16.  One wave32 per 16-row tile.
//  - A-fragments gathered directly from global (16B chunks match the 16-bit
//    A layout), software-pipelined one kb ahead.
//  - B (weights) packed kb-major per-launch; each kb-slice staged into LDS
//    once per WORKGROUP (4 waves share it) with GLOBAL_LOAD_ASYNC_TO_LDS_B128
//    double buffering (ASYNCcnt), falling back to sync staging if the builtin
//    is unavailable.  B-fragment feeds are ping-ponged ds_load_b128.
//  - LayerNorm in registers on the WMMA C layout via 16-lane shfl_xor.
//  - Scatter-mean fused into the edge kernel with global_atomic_add_f32.
// ---------------------------------------------------------------------------

#define NN 50000
#define NE 600000
#define HD 128
#define NL 10
#define LN_EPS 1e-5f

typedef __attribute__((ext_vector_type(16))) __bf16 v16bf;
typedef __attribute__((ext_vector_type(8))) float v8f;

union BFrag {
  v16bf v;
  float4 f4[2];
  unsigned short s[16];
};

#if defined(__has_builtin)
#if __has_builtin(__builtin_amdgcn_global_load_async_to_lds_b128)
#define HAVE_ASYNC_LDS 1
#endif
#endif
#ifndef HAVE_ASYNC_LDS
#define HAVE_ASYNC_LDS 0
#endif

#if HAVE_ASYNC_LDS
typedef int v4i __attribute__((vector_size(16)));
typedef __attribute__((address_space(1))) v4i gv4i;  // global int4
typedef __attribute__((address_space(3))) v4i sv4i;  // LDS int4
#endif

template <int N>
__device__ __forceinline__ void async_wait() {
#if HAVE_ASYNC_LDS
  asm volatile("s_wait_asynccnt %0" ::"i"(N) : "memory");
#endif
}

// Stage CHUNKS 16-byte chunks from global to LDS, distributed over 128 thr.
// CHUNKS is a compile-time multiple of 128 -> branch-free async clause.
template <int CHUNKS>
__device__ __forceinline__ void stage_slice(const unsigned short* __restrict__ gsrc,
                                            unsigned short* ldst, int tid) {
#if HAVE_ASYNC_LDS
#pragma unroll
  for (int k = 0; k < CHUNKS / 128; ++k) {
    const size_t i = (size_t)(k * 128 + tid) * 8;
    __builtin_amdgcn_global_load_async_to_lds_b128(
        (gv4i*)(gsrc + i), (sv4i*)(ldst + i), 0, 0);
  }
#else
#pragma unroll
  for (int k = 0; k < CHUNKS / 128; ++k) {
    const size_t i = (size_t)(k * 128 + tid) * 8;
    *(float4*)(ldst + i) = *(const float4*)(gsrc + i);
  }
#endif
}

__device__ __forceinline__ void ld_bfrag(BFrag& b, const unsigned short* bs,
                                         int lane, int nb) {
  const unsigned short* p = bs + nb * 512 + lane * 16;
  b.f4[0] = *(const float4*)(p);
  b.f4[1] = *(const float4*)(p + 8);
}

__device__ __forceinline__ unsigned short f2bf(float x) {
  unsigned int u = __float_as_uint(x);
  u += 0x7FFFu + ((u >> 16) & 1u);  // round-to-nearest-even
  return (unsigned short)(u >> 16);
}

__device__ __forceinline__ float gelu_exact(float x) {
  return 0.5f * x * (1.0f + erff(x * 0.70710678118654752f));
}

// reduce across the 16 lanes holding one matrix row (halves stay separate)
__device__ __forceinline__ float hsum16(float v) {
  v += __shfl_xor(v, 1, 32);
  v += __shfl_xor(v, 2, 32);
  v += __shfl_xor(v, 4, 32);
  v += __shfl_xor(v, 8, 32);
  return v;
}
__device__ __forceinline__ float hsum32(float v) {
  v = hsum16(v);
  v += __shfl_xor(v, 16, 32);
  return v;
}

// ---------------------------------------------------------------------------
// Weight packing: row-major fp32 [K x Nsrc] -> bf16 B-fragments, kb-major so
// one kb-slice (all nb) is contiguous for async staging.
// Fragment f = kb*NB + nb; lane holds 16 contiguous bf16 (32B):
// lanes 0-15 hold K=kb*32+{0..7,16..23} of column n=nb*16+lane%16, lanes 16-31 K={8..15,24..31}.
// ---------------------------------------------------------------------------
__global__ void pack_b_kernel(const float* __restrict__ src,
                              unsigned short* __restrict__ dst,
                              int K, int Nsrc, int Npad) {
  int idx = blockIdx.x * 256 + threadIdx.x;
  if (idx >= K * Npad) return;
  int k = idx / Npad, n = idx % Npad;
  float val = (n < Nsrc) ? src[(size_t)k * Nsrc + n] : 0.f;
  int kb = k >> 5, kk = k & 31;
  int hi = (kk >> 3) & 1;
  int j = (kk & 7) + ((kk >> 4) << 3);
  int lane = (n & 15) + (hi << 4);
  int nb = n >> 4;
  int NB = Npad >> 4;
  dst[((size_t)(kb * NB + nb) * 32 + lane) * 16 + j] = f2bf(val);
}

__global__ void zero_f32_kernel(float* __restrict__ p, int n) {
  int i = blockIdx.x * 256 + threadIdx.x;
  if (i < n) p[i] = 0.f;
}

__global__ void degree_kernel(const int* __restrict__ col, float* __restrict__ deg) {
  int e = blockIdx.x * 256 + threadIdx.x;
  if (e < NE) atomicAdd(&deg[col[e]], 1.0f);
}

__global__ void finalize_agg_kernel(const float* __restrict__ agg,
                                    const float* __restrict__ deg,
                                    unsigned short* __restrict__ aggb) {
  int i = blockIdx.x * 256 + threadIdx.x;
  if (i < NN * HD) {
    float d = fmaxf(deg[i >> 7], 1.0f);
    aggb[i] = f2bf(agg[i] / d);
  }
}

// ---------------------------------------------------------------------------
// Encoder: h = gelu(LN(x @ W + b)); one wave per node, 4 features/lane.
// ---------------------------------------------------------------------------
__global__ __launch_bounds__(128) void encoder_kernel(
    const float* __restrict__ x, const float* __restrict__ W,
    const float* __restrict__ b, const float* __restrict__ g,
    const float* __restrict__ beta, float* __restrict__ hf,
    unsigned short* __restrict__ hbf) {
  const int lane = threadIdx.x & 31, wid = threadIdx.x >> 5;
  const int node = blockIdx.x * 4 + wid;
  if (node >= NN) return;
  float xv[7];
#pragma unroll
  for (int k = 0; k < 7; ++k) xv[k] = x[(size_t)node * 7 + k];
  float a[4];
#pragma unroll
  for (int j = 0; j < 4; ++j) {
    const int f = lane * 4 + j;
    float s = b[f];
#pragma unroll
    for (int k = 0; k < 7; ++k) s = fmaf(xv[k], W[k * HD + f], s);
    a[j] = s;
  }
  float s1 = 0.f, s2 = 0.f;
#pragma unroll
  for (int j = 0; j < 4; ++j) { s1 += a[j]; s2 += a[j] * a[j]; }
  const float mu = hsum32(s1) * (1.f / 128.f);
  const float ex2 = hsum32(s2) * (1.f / 128.f);
  const float rstd = rsqrtf(fmaxf(ex2 - mu * mu, 0.f) + LN_EPS);
#pragma unroll
  for (int j = 0; j < 4; ++j) {
    const int f = lane * 4 + j;
    float v = (a[j] - mu) * rstd * g[f] + beta[f];
    v = gelu_exact(v);
    hf[(size_t)node * HD + f] = v;
    hbf[(size_t)node * HD + f] = f2bf(v);
  }
}

// Edge encoder: ea = edge_attr @ W + b (no LN/GELU).
__global__ __launch_bounds__(128) void edge_encoder_kernel(
    const float* __restrict__ ea, const float* __restrict__ W,
    const float* __restrict__ b, float* __restrict__ eaf,
    unsigned short* __restrict__ eabf) {
  const int lane = threadIdx.x & 31, wid = threadIdx.x >> 5;
  const int e = blockIdx.x * 4 + wid;
  if (e >= NE) return;
  float xv[8];
#pragma unroll
  for (int k = 0; k < 8; ++k) xv[k] = ea[(size_t)e * 8 + k];
#pragma unroll
  for (int j = 0; j < 4; ++j) {
    const int f = lane * 4 + j;
    float s = b[f];
#pragma unroll
    for (int k = 0; k < 8; ++k) s = fmaf(xv[k], W[k * HD + f], s);
    eaf[(size_t)e * HD + f] = s;
    eabf[(size_t)e * HD + f] = f2bf(s);
  }
}

// ---------------------------------------------------------------------------
// Edge update: [h[row]|h[col]|ea] (16x384) @ w1 -> LN,GELU -> @ w2 -> LN,
// residual into ea (fp32+bf16), fused atomic scatter-add into agg.
// One wave per 16 edges; 192 + 64 WMMAs per tile; block stages B in LDS.
// ---------------------------------------------------------------------------
__global__ __launch_bounds__(128, 1) void edge_update_kernel(
    const unsigned short* __restrict__ hbf, unsigned short* __restrict__ eabf,
    float* __restrict__ eaf, float* __restrict__ agg,
    const int* __restrict__ row, const int* __restrict__ col,
    const unsigned short* __restrict__ w1p, const float* __restrict__ b1,
    const float* __restrict__ g1, const float* __restrict__ bt1,
    const unsigned short* __restrict__ w2p, const float* __restrict__ b2,
    const float* __restrict__ g2, const float* __restrict__ bt2) {
  __shared__ unsigned short bstage[2][16 * 512];  // 2 x 16KB double buffer
  __shared__ unsigned short hid[4][16 * 256];     // 32KB hidden re-layout
  const int tid = threadIdx.x;
  const int lane = tid & 31;
  const int wid = tid >> 5;
  int tile = blockIdx.x * 4 + wid;
  const bool active = (tile * 16 < NE);
  if (!active) tile = (NE / 16) - 1;
  const int r = lane & 15;
  const int hi = lane >> 4;

  int e = tile * 16 + r;
  if (e >= NE) e = NE - 1;
  const int re = row[e];
  const int ce = col[e];
  const unsigned short* seg0 = hbf + (size_t)re * HD;
  const unsigned short* seg1 = hbf + (size_t)ce * HD;
  const unsigned short* seg2 = eabf + (size_t)e * HD;

  auto load_a1 = [&](int kb) {
    const int base = kb * 32 + hi * 8;
    const unsigned short* sp = (base < 128) ? seg0 : ((base < 256) ? seg1 : seg2);
    const int o = base & 127;
    BFrag a;
    a.f4[0] = *(const float4*)(sp + o);
    a.f4[1] = *(const float4*)(sp + o + 16);
    return a;
  };

  const v8f zero8 = {0.f, 0.f, 0.f, 0.f, 0.f, 0.f, 0.f, 0.f};
  v8f acc[16];
#pragma unroll
  for (int nb = 0; nb < 16; ++nb) acc[nb] = zero8;

  // ---- GEMM1: 16x384 @ 384x256, B double-buffered through LDS ----
  stage_slice<1024>(w1p, bstage[0], tid);
  BFrag a_cur = load_a1(0);
#pragma unroll
  for (int kb = 0; kb < 12; ++kb) {
    const int nxt = kb + 1;
    if (nxt < 12) {
      stage_slice<1024>(w1p + (size_t)nxt * 16 * 512, bstage[nxt & 1], tid);
      async_wait<8>();  // current slice done; next slice may stay in flight
    } else {
      async_wait<0>();
    }
    __syncthreads();
    BFrag a_nxt = (nxt < 12) ? load_a1(nxt) : a_cur;  // prefetch next A
    const unsigned short* bs = bstage[kb & 1];
    BFrag bb[2];
    ld_bfrag(bb[0], bs, lane, 0);
#pragma unroll
    for (int nb = 0; nb < 16; ++nb) {
      if (nb + 1 < 16) ld_bfrag(bb[(nb + 1) & 1], bs, lane, nb + 1);
      acc[nb] = __builtin_amdgcn_wmma_f32_16x16x32_bf16(
          false, a_cur.v, false, bb[nb & 1].v, (short)0, acc[nb], false, false);
    }
    __syncthreads();  // protect buffer being overwritten next iteration
    a_cur = a_nxt;
  }

  // bias + LayerNorm over 256 hidden features (row stats via half-wave shfl)
  float mean1[8], rstd1[8];
  {
    float s1[8], s2[8];
#pragma unroll
    for (int v = 0; v < 8; ++v) { s1[v] = 0.f; s2[v] = 0.f; }
#pragma unroll
    for (int nb = 0; nb < 16; ++nb) {
      const float bv = b1[nb * 16 + r];
#pragma unroll
      for (int v = 0; v < 8; ++v) {
        const float xv = acc[nb][v] + bv;
        acc[nb][v] = xv;
        s1[v] += xv;
        s2[v] += xv * xv;
      }
    }
#pragma unroll
    for (int v = 0; v < 8; ++v) {
      const float mu = hsum16(s1[v]) * (1.f / 256.f);
      const float ex2 = hsum16(s2[v]) * (1.f / 256.f);
      mean1[v] = mu;
      rstd1[v] = rsqrtf(fmaxf(ex2 - mu * mu, 0.f) + LN_EPS);
    }
  }

  unsigned short* hrow = hid[wid];
#pragma unroll
  for (int nb = 0; nb < 16; ++nb) {
    const float gv = g1[nb * 16 + r];
    const float bv = bt1[nb * 16 + r];
#pragma unroll
    for (int v = 0; v < 8; ++v) {
      const int m = v + 8 * hi;
      float xv = (acc[nb][v] - mean1[v]) * rstd1[v] * gv + bv;
      xv = gelu_exact(xv);
      hrow[m * 256 + nb * 16 + r] = f2bf(xv);
    }
  }

  auto load_a2 = [&](int kb) {
    const int base = kb * 32 + hi * 8;
    BFrag a;
    a.f4[0] = *(const float4*)(hrow + r * 256 + base);
    a.f4[1] = *(const float4*)(hrow + r * 256 + base + 16);
    return a;
  };

  v8f acc2[8];
#pragma unroll
  for (int nb = 0; nb < 8; ++nb) acc2[nb] = zero8;

  // ---- GEMM2: 16x256 @ 256x128 ----
  stage_slice<512>(w2p, bstage[0], tid);
  BFrag a2 = load_a2(0);
#pragma unroll
  for (int kb = 0; kb < 8; ++kb) {
    const int nxt = kb + 1;
    if (nxt < 8) {
      stage_slice<512>(w2p + (size_t)nxt * 8 * 512, bstage[nxt & 1], tid);
      async_wait<4>();
    } else {
      async_wait<0>();
    }
    __syncthreads();
    BFrag a_nxt = (nxt < 8) ? load_a2(nxt) : a2;
    const unsigned short* bs = bstage[kb & 1];
    BFrag bb[2];
    ld_bfrag(bb[0], bs, lane, 0);
#pragma unroll
    for (int nb = 0; nb < 8; ++nb) {
      if (nb + 1 < 8) ld_bfrag(bb[(nb + 1) & 1], bs, lane, nb + 1);
      acc2[nb] = __builtin_amdgcn_wmma_f32_16x16x32_bf16(
          false, a2.v, false, bb[nb & 1].v, (short)0, acc2[nb], false, false);
    }
    __syncthreads();
    a2 = a_nxt;
  }

  float mean2[8], rstd2[8];
  {
    float s1[8], s2[8];
#pragma unroll
    for (int v = 0; v < 8; ++v) { s1[v] = 0.f; s2[v] = 0.f; }
#pragma unroll
    for (int nb = 0; nb < 8; ++nb) {
      const float bv = b2[nb * 16 + r];
#pragma unroll
      for (int v = 0; v < 8; ++v) {
        const float xv = acc2[nb][v] + bv;
        acc2[nb][v] = xv;
        s1[v] += xv;
        s2[v] += xv * xv;
      }
    }
#pragma unroll
    for (int v = 0; v < 8; ++v) {
      const float mu = hsum16(s1[v]) * (1.f / 128.f);
      const float ex2 = hsum16(s2[v]) * (1.f / 128.f);
      mean2[v] = mu;
      rstd2[v] = rsqrtf(fmaxf(ex2 - mu * mu, 0.f) + LN_EPS);
    }
  }

  int ecol[8];
#pragma unroll
  for (int v = 0; v < 8; ++v) {
    int ee = tile * 16 + v + 8 * hi;
    ecol[v] = col[(ee < NE) ? ee : (NE - 1)];
  }

#pragma unroll
  for (int nb = 0; nb < 8; ++nb) {
    const float gv = g2[nb * 16 + r];
    const float bv = bt2[nb * 16 + r];
#pragma unroll
    for (int v = 0; v < 8; ++v) {
      const int ee = tile * 16 + v + 8 * hi;
      if (active && ee < NE) {
        const int n = nb * 16 + r;
        const float xv = (acc2[nb][v] - mean2[v]) * rstd2[v] * gv + bv;
        const size_t idx = (size_t)ee * HD + n;
        const float nv = eaf[idx] + xv;
        eaf[idx] = nv;
        eabf[idx] = f2bf(nv);
        atomicAdd(agg + (size_t)ecol[v] * HD + n, nv);
      }
    }
  }
}

// ---------------------------------------------------------------------------
// Node update: [h|agg] (16x256) @ mw1 -> LN,GELU -> @ mw2 -> LN, residual h.
// ---------------------------------------------------------------------------
__global__ __launch_bounds__(128, 1) void node_update_kernel(
    float* __restrict__ hf, unsigned short* __restrict__ hbf,
    const unsigned short* __restrict__ aggb,
    const unsigned short* __restrict__ w1p, const float* __restrict__ b1,
    const float* __restrict__ g1, const float* __restrict__ bt1,
    const unsigned short* __restrict__ w2p, const float* __restrict__ b2,
    const float* __restrict__ g2, const float* __restrict__ bt2) {
  __shared__ unsigned short bstage[2][16 * 512];
  __shared__ unsigned short hid[4][16 * 256];
  const int tid = threadIdx.x;
  const int lane = tid & 31;
  const int wid = tid >> 5;
  int tile = blockIdx.x * 4 + wid;
  const bool active = (tile * 16 < NN);
  if (!active) tile = (NN / 16) - 1;
  const int r = lane & 15;
  const int hi = lane >> 4;

  int nd = tile * 16 + r;
  if (nd >= NN) nd = NN - 1;
  const unsigned short* seg0 = hbf + (size_t)nd * HD;
  const unsigned short* seg1 = aggb + (size_t)nd * HD;

  auto load_a1 = [&](int kb) {
    const int base = kb * 32 + hi * 8;
    const unsigned short* sp = (base < 128) ? seg0 : seg1;
    const int o = base & 127;
    BFrag a;
    a.f4[0] = *(const float4*)(sp + o);
    a.f4[1] = *(const float4*)(sp + o + 16);
    return a;
  };

  const v8f zero8 = {0.f, 0.f, 0.f, 0.f, 0.f, 0.f, 0.f, 0.f};
  v8f acc[16];
#pragma unroll
  for (int nb = 0; nb < 16; ++nb) acc[nb] = zero8;

  stage_slice<1024>(w1p, bstage[0], tid);
  BFrag a_cur = load_a1(0);
#pragma unroll
  for (int kb = 0; kb < 8; ++kb) {
    const int nxt = kb + 1;
    if (nxt < 8) {
      stage_slice<1024>(w1p + (size_t)nxt * 16 * 512, bstage[nxt & 1], tid);
      async_wait<8>();
    } else {
      async_wait<0>();
    }
    __syncthreads();
    BFrag a_nxt = (nxt < 8) ? load_a1(nxt) : a_cur;
    const unsigned short* bs = bstage[kb & 1];
    BFrag bb[2];
    ld_bfrag(bb[0], bs, lane, 0);
#pragma unroll
    for (int nb = 0; nb < 16; ++nb) {
      if (nb + 1 < 16) ld_bfrag(bb[(nb + 1) & 1], bs, lane, nb + 1);
      acc[nb] = __builtin_amdgcn_wmma_f32_16x16x32_bf16(
          false, a_cur.v, false, bb[nb & 1].v, (short)0, acc[nb], false, false);
    }
    __syncthreads();
    a_cur = a_nxt;
  }

  float mean1[8], rstd1[8];
  {
    float s1[8], s2[8];
#pragma unroll
    for (int v = 0; v < 8; ++v) { s1[v] = 0.f; s2[v] = 0.f; }
#pragma unroll
    for (int nb = 0; nb < 16; ++nb) {
      const float bv = b1[nb * 16 + r];
#pragma unroll
      for (int v = 0; v < 8; ++v) {
        const float xv = acc[nb][v] + bv;
        acc[nb][v] = xv;
        s1[v] += xv;
        s2[v] += xv * xv;
      }
    }
#pragma unroll
    for (int v = 0; v < 8; ++v) {
      const float mu = hsum16(s1[v]) * (1.f / 256.f);
      const float ex2 = hsum16(s2[v]) * (1.f / 256.f);
      mean1[v] = mu;
      rstd1[v] = rsqrtf(fmaxf(ex2 - mu * mu, 0.f) + LN_EPS);
    }
  }

  unsigned short* hrow = hid[wid];
#pragma unroll
  for (int nb = 0; nb < 16; ++nb) {
    const float gv = g1[nb * 16 + r];
    const float bv = bt1[nb * 16 + r];
#pragma unroll
    for (int v = 0; v < 8; ++v) {
      const int m = v + 8 * hi;
      float xv = (acc[nb][v] - mean1[v]) * rstd1[v] * gv + bv;
      xv = gelu_exact(xv);
      hrow[m * 256 + nb * 16 + r] = f2bf(xv);
    }
  }

  auto load_a2 = [&](int kb) {
    const int base = kb * 32 + hi * 8;
    BFrag a;
    a.f4[0] = *(const float4*)(hrow + r * 256 + base);
    a.f4[1] = *(const float4*)(hrow + r * 256 + base + 16);
    return a;
  };

  v8f acc2[8];
#pragma unroll
  for (int nb = 0; nb < 8; ++nb) acc2[nb] = zero8;

  stage_slice<512>(w2p, bstage[0], tid);
  BFrag a2 = load_a2(0);
#pragma unroll
  for (int kb = 0; kb < 8; ++kb) {
    const int nxt = kb + 1;
    if (nxt < 8) {
      stage_slice<512>(w2p + (size_t)nxt * 8 * 512, bstage[nxt & 1], tid);
      async_wait<4>();
    } else {
      async_wait<0>();
    }
    __syncthreads();
    BFrag a_nxt = (nxt < 8) ? load_a2(nxt) : a2;
    const unsigned short* bs = bstage[kb & 1];
    BFrag bb[2];
    ld_bfrag(bb[0], bs, lane, 0);
#pragma unroll
    for (int nb = 0; nb < 8; ++nb) {
      if (nb + 1 < 8) ld_bfrag(bb[(nb + 1) & 1], bs, lane, nb + 1);
      acc2[nb] = __builtin_amdgcn_wmma_f32_16x16x32_bf16(
          false, a2.v, false, bb[nb & 1].v, (short)0, acc2[nb], false, false);
    }
    __syncthreads();
    a2 = a_nxt;
  }

  float mean2[8], rstd2[8];
  {
    float s1[8], s2[8];
#pragma unroll
    for (int v = 0; v < 8; ++v) { s1[v] = 0.f; s2[v] = 0.f; }
#pragma unroll
    for (int nb = 0; nb < 8; ++nb) {
      const float bv = b2[nb * 16 + r];
#pragma unroll
      for (int v = 0; v < 8; ++v) {
        const float xv = acc2[nb][v] + bv;
        acc2[nb][v] = xv;
        s1[v] += xv;
        s2[v] += xv * xv;
      }
    }
#pragma unroll
    for (int v = 0; v < 8; ++v) {
      const float mu = hsum16(s1[v]) * (1.f / 128.f);
      const float ex2 = hsum16(s2[v]) * (1.f / 128.f);
      mean2[v] = mu;
      rstd2[v] = rsqrtf(fmaxf(ex2 - mu * mu, 0.f) + LN_EPS);
    }
  }

#pragma unroll
  for (int nb = 0; nb < 8; ++nb) {
    const float gv = g2[nb * 16 + r];
    const float bv = bt2[nb * 16 + r];
#pragma unroll
    for (int v = 0; v < 8; ++v) {
      const int node2 = tile * 16 + v + 8 * hi;
      if (active && node2 < NN) {
        const int n = nb * 16 + r;
        const float xv = (acc2[nb][v] - mean2[v]) * rstd2[v] * gv + bv;
        const size_t idx = (size_t)node2 * HD + n;
        const float nv = hf[idx] + xv;
        hf[idx] = nv;
        hbf[idx] = f2bf(nv);
      }
    }
  }
}

// ---------------------------------------------------------------------------
// Decoder: gelu(h @ W1 + b1) @ W2 + b2 -> out (N,4).  W2 zero-padded to 16 cols.
// Small: B direct from L2, no staging (no barriers, early-exit OK).
// ---------------------------------------------------------------------------
__global__ __launch_bounds__(128, 1) void decoder_kernel(
    const unsigned short* __restrict__ hbf,
    const unsigned short* __restrict__ dW1p, const float* __restrict__ db1,
    const unsigned short* __restrict__ dW2p, const float* __restrict__ db2,
    float* __restrict__ out) {
  __shared__ unsigned short hid[4][16 * 128];
  const int lane = threadIdx.x & 31;
  const int wid = threadIdx.x >> 5;
  const int tile = blockIdx.x * 4 + wid;
  if (tile * 16 >= NN) return;
  const int r = lane & 15;
  const int hi = lane >> 4;

  int nd = tile * 16 + r;
  if (nd >= NN) nd = NN - 1;
  const unsigned short* seg0 = hbf + (size_t)nd * HD;

  const v8f zero8 = {0.f, 0.f, 0.f, 0.f, 0.f, 0.f, 0.f, 0.f};
  v8f acc[8];
#pragma unroll
  for (int nb = 0; nb < 8; ++nb) acc[nb] = zero8;

#pragma unroll
  for (int kb = 0; kb < 4; ++kb) {
    const int base = kb * 32 + hi * 8;
    BFrag a;
    a.f4[0] = *(const float4*)(seg0 + base);
    a.f4[1] = *(const float4*)(seg0 + base + 16);
#pragma unroll
    for (int nb = 0; nb < 8; ++nb) {
      BFrag b;
      const unsigned short* bp = dW1p + ((size_t)(kb * 8 + nb) * 32 + lane) * 16;
      b.f4[0] = *(const float4*)(bp);
      b.f4[1] = *(const float4*)(bp + 8);
      acc[nb] = __builtin_amdgcn_wmma_f32_16x16x32_bf16(
          false, a.v, false, b.v, (short)0, acc[nb], false, false);
    }
  }

  unsigned short* hrow = hid[wid];
#pragma unroll
  for (int nb = 0; nb < 8; ++nb) {
    const float bv = db1[nb * 16 + r];
#pragma unroll
    for (int v = 0; v < 8; ++v) {
      const int m = v + 8 * hi;
      hrow[m * 128 + nb * 16 + r] = f2bf(gelu_exact(acc[nb][v] + bv));
    }
  }

  v8f acc2 = zero8;
#pragma unroll
  for (int kb = 0; kb < 4; ++kb) {
    const int base = kb * 32 + hi * 8;
    BFrag a;
    a.f4[0] = *(const float4*)(hrow + r * 128 + base);
    a.f4[1] = *(const float4*)(hrow + r * 128 + base + 16);
    BFrag b;
    const unsigned short* bp = dW2p + ((size_t)kb * 32 + lane) * 16;
    b.f4[0] = *(const float4*)(bp);
    b.f4[1] = *(const float4*)(bp + 8);
    acc2 = __builtin_amdgcn_wmma_f32_16x16x32_bf16(
        false, a.v, false, b.v, (short)0, acc2, false, false);
  }

  const float bv2 = (r < 4) ? db2[r] : 0.f;
#pragma unroll
  for (int v = 0; v < 8; ++v) {
    const int node = tile * 16 + v + 8 * hi;
    if (node < NN && r < 4) out[(size_t)node * 4 + r] = acc2[v] + bv2;
  }
}

// ---------------------------------------------------------------------------
extern "C" void kernel_launch(void* const* d_in, const int* in_sizes, int n_in,
                              void* d_out, int out_size, void* d_ws, size_t ws_size,
                              hipStream_t stream) {
  (void)in_sizes; (void)n_in; (void)out_size; (void)ws_size;
  const float* x      = (const float*)d_in[0];
  const int*   ei     = (const int*)d_in[1];
  const float* eattr  = (const float*)d_in[2];
  const float* enc_W  = (const float*)d_in[3];
  const float* enc_b  = (const float*)d_in[4];
  const float* enc_g  = (const float*)d_in[5];
  const float* enc_bt = (const float*)d_in[6];
  const float* ee_W   = (const float*)d_in[7];
  const float* ee_b   = (const float*)d_in[8];
  const float* eW1  = (const float*)d_in[9];
  const float* eb1  = (const float*)d_in[10];
  const float* eg1  = (const float*)d_in[11];
  const float* ebt1 = (const float*)d_in[12];
  const float* eW2  = (const float*)d_in[13];
  const float* eb2  = (const float*)d_in[14];
  const float* eg2  = (const float*)d_in[15];
  const float* ebt2 = (const float*)d_in[16];
  const float* nW1  = (const float*)d_in[17];
  const float* nb1  = (const float*)d_in[18];
  const float* ng1  = (const float*)d_in[19];
  const float* nbt1 = (const float*)d_in[20];
  const float* nW2  = (const float*)d_in[21];
  const float* nb2  = (const float*)d_in[22];
  const float* ng2  = (const float*)d_in[23];
  const float* nbt2 = (const float*)d_in[24];
  const float* dW1  = (const float*)d_in[25];
  const float* db1  = (const float*)d_in[26];
  const float* dW2  = (const float*)d_in[27];
  const float* db2  = (const float*)d_in[28];
  float* out = (float*)d_out;

  const int* row = ei;
  const int* col = ei + NE;

  // --- carve workspace (256B aligned regions) ---
  char* ws = (char*)d_ws;
  size_t off = 0;
  auto carve = [&](size_t bytes) {
    char* p = ws + off;
    off = (off + bytes + 255) & ~(size_t)255;
    return p;
  };
  float*          hf    = (float*)carve((size_t)NN * HD * 4);
  unsigned short* hbf   = (unsigned short*)carve((size_t)NN * HD * 2);
  float*          eaf   = (float*)carve((size_t)NE * HD * 4);
  unsigned short* eabf  = (unsigned short*)carve((size_t)NE * HD * 2);
  float*          agg   = (float*)carve((size_t)NN * HD * 4);
  unsigned short* aggb  = (unsigned short*)carve((size_t)NN * HD * 2);
  float*          deg   = (float*)carve((size_t)NN * 4);
  unsigned short* w1p   = (unsigned short*)carve((size_t)NL * 384 * 256 * 2);
  unsigned short* w2p   = (unsigned short*)carve((size_t)NL * 256 * 128 * 2);
  unsigned short* mw1p  = (unsigned short*)carve((size_t)NL * 256 * 256 * 2);
  unsigned short* mw2p  = (unsigned short*)carve((size_t)NL * 256 * 128 * 2);
  unsigned short* dW1p  = (unsigned short*)carve((size_t)128 * 128 * 2);
  unsigned short* dW2p  = (unsigned short*)carve((size_t)128 * 16 * 2);

  // --- pack weights into WMMA B-fragment layout (bf16, kb-major) ---
  for (int l = 0; l < NL; ++l) {
    pack_b_kernel<<<(384 * 256 + 255) / 256, 256, 0, stream>>>(
        eW1 + (size_t)l * 384 * 256, w1p + (size_t)l * 384 * 256, 384, 256, 256);
    pack_b_kernel<<<(256 * 128 + 255) / 256, 256, 0, stream>>>(
        eW2 + (size_t)l * 256 * 128, w2p + (size_t)l * 256 * 128, 256, 128, 128);
    pack_b_kernel<<<(256 * 256 + 255) / 256, 256, 0, stream>>>(
        nW1 + (size_t)l * 256 * 256, mw1p + (size_t)l * 256 * 256, 256, 256, 256);
    pack_b_kernel<<<(256 * 128 + 255) / 256, 256, 0, stream>>>(
        nW2 + (size_t)l * 256 * 128, mw2p + (size_t)l * 256 * 128, 256, 128, 128);
  }
  pack_b_kernel<<<(128 * 128 + 255) / 256, 256, 0, stream>>>(dW1, dW1p, 128, 128, 128);
  pack_b_kernel<<<(128 * 16 + 255) / 256, 256, 0, stream>>>(dW2, dW2p, 128, 4, 16);

  // --- degrees (same every layer) ---
  zero_f32_kernel<<<(NN + 255) / 256, 256, 0, stream>>>(deg, NN);
  degree_kernel<<<(NE + 255) / 256, 256, 0, stream>>>(col, deg);

  // --- encoders ---
  encoder_kernel<<<(NN + 3) / 4, 128, 0, stream>>>(x, enc_W, enc_b, enc_g, enc_bt, hf, hbf);
  edge_encoder_kernel<<<(NE + 3) / 4, 128, 0, stream>>>(eattr, ee_W, ee_b, eaf, eabf);

  // --- message passing layers ---
  const int edge_blocks = ((NE + 15) / 16 + 3) / 4;
  const int node_blocks = ((NN + 15) / 16 + 3) / 4;
  for (int l = 0; l < NL; ++l) {
    zero_f32_kernel<<<(NN * HD + 255) / 256, 256, 0, stream>>>(agg, NN * HD);
    edge_update_kernel<<<edge_blocks, 128, 0, stream>>>(
        hbf, eabf, eaf, agg, row, col,
        w1p + (size_t)l * 384 * 256, eb1 + (size_t)l * 256,
        eg1 + (size_t)l * 256, ebt1 + (size_t)l * 256,
        w2p + (size_t)l * 256 * 128, eb2 + (size_t)l * 128,
        eg2 + (size_t)l * 128, ebt2 + (size_t)l * 128);
    finalize_agg_kernel<<<(NN * HD + 255) / 256, 256, 0, stream>>>(agg, deg, aggb);
    node_update_kernel<<<node_blocks, 128, 0, stream>>>(
        hf, hbf, aggb,
        mw1p + (size_t)l * 256 * 256, nb1 + (size_t)l * 256,
        ng1 + (size_t)l * 256, nbt1 + (size_t)l * 256,
        mw2p + (size_t)l * 256 * 128, nb2 + (size_t)l * 128,
        ng2 + (size_t)l * 128, nbt2 + (size_t)l * 128);
  }

  // --- decoder ---
  decoder_kernel<<<node_blocks, 128, 0, stream>>>(hbf, dW1p, db1, dW2p, db2, out);
}
